// ClassicalSelfAttention_65481071404657
// MI455X (gfx1250) — matrix operand order repaired and checked by reading, not compiled, not run on hardware
//
#include <hip/hip_runtime.h>
#include <hip/hip_bf16.h>

typedef __bf16 bf16_t;
typedef bf16_t v16bf __attribute__((ext_vector_type(16)));
typedef bf16_t v8bf  __attribute__((ext_vector_type(8)));
typedef float  v8f   __attribute__((ext_vector_type(8)));
typedef int    v4i   __attribute__((ext_vector_type(4)));

#define Bq 4
#define Nn 4096
#define Dd 1024
#define QSCALE 0.03125f  // 1/sqrt(1024)

// LDS tile geometry (padded for conflict-free ds_load_b128)
#define KROW 1032                    // 1024 dims + 8 pad (2064B rows, 16B aligned)
#define XROW 40                      // 32 keys + 8 pad (80B rows, 16B aligned)
#define KTILE_B (32 * KROW * 2)      // 66048 bytes
#define XTILE_B (1024 * XROW * 2)    // 81920 bytes
#define SMEM_BYTES (2 * KTILE_B + 2 * XTILE_B + 1024 * 4 + 64 * 4)

// ---- CDNA5 async-to-LDS path (guarded; falls back to plain LDS stores) ----
#ifndef HAVE_ASYNC
#if defined(__has_builtin)
#if __has_builtin(__builtin_amdgcn_global_load_async_to_lds_b128) && \
    __has_builtin(__builtin_amdgcn_s_wait_asynccnt)
#define HAVE_ASYNC 1
#endif
#endif
#endif
#ifndef HAVE_ASYNC
#define HAVE_ASYNC 0
#endif

static __device__ __forceinline__ void async_copy16(const void* g, void* l) {
#if HAVE_ASYNC
    __builtin_amdgcn_global_load_async_to_lds_b128(
        (__attribute__((address_space(1))) v4i*)g,
        (__attribute__((address_space(3))) v4i*)l, 0, 0);
#else
    *(uint4*)l = *(const uint4*)g;
#endif
}
static __device__ __forceinline__ void async_wait_all() {
#if HAVE_ASYNC
    __builtin_amdgcn_s_wait_asynccnt(0);
#endif
}

static __device__ __forceinline__ unsigned short f32_to_bf16_bits(float f) {
    unsigned int u = __builtin_bit_cast(unsigned int, f);
    unsigned int rounding = 0x7FFFu + ((u >> 16) & 1u);
    u += rounding;
    return (unsigned short)(u >> 16);
}

static __device__ __forceinline__ v16bf ld16(const bf16_t* p0, const bf16_t* p1) {
    union { v16bf v; v8bf h[2]; } u;
    u.h[0] = *(const v8bf*)p0;
    u.h[1] = *(const v8bf*)p1;
    return u.v;
}

static __device__ __forceinline__ v8f wmma_bf16(v16bf a, v16bf b, v8f c) {
    return __builtin_amdgcn_wmma_f32_16x16x32_bf16(false, a, false, b, (short)0, c, false, false);
}

// ---------------- conversion kernels ----------------
__global__ __launch_bounds__(256) void cvt_x_kernel(const float* __restrict__ x,
                                                    unsigned short* __restrict__ xb,
                                                    unsigned short* __restrict__ xt) {
    size_t i = (size_t)blockIdx.x * 256 + threadIdx.x;   // over B*N*D
    float v = x[i];
    unsigned short h = f32_to_bf16_bits(v);
    xb[i] = h;
    int d = (int)(i & (Dd - 1));
    size_t bn = i >> 10;
    int n = (int)(bn & (Nn - 1));
    int b = (int)(bn >> 12);
    xt[((size_t)b * Dd + d) * Nn + n] = h;               // xT: [B][D][N]
}

__global__ __launch_bounds__(256) void cvt_w_kernel(const float* __restrict__ Wq,
                                                    const float* __restrict__ Wk,
                                                    unsigned short* __restrict__ Wqb,
                                                    unsigned short* __restrict__ Wkb) {
    size_t i = (size_t)blockIdx.x * 256 + threadIdx.x;   // over D*D
    Wqb[i] = f32_to_bf16_bits(Wq[i]);
    Wkb[i] = f32_to_bf16_bits(Wk[i]);
}

// ---------------- projection GEMM: q = (x Wq^T)*scale, k = x Wk^T ----------------
__global__ __launch_bounds__(256) void proj_kernel(const unsigned short* __restrict__ xb,
                                                   const unsigned short* __restrict__ Wqb,
                                                   const unsigned short* __restrict__ Wkb,
                                                   unsigned short* __restrict__ qb,
                                                   unsigned short* __restrict__ kb) {
    const unsigned short* W   = blockIdx.y ? Wkb : Wqb;
    unsigned short*       dst = blockIdx.y ? kb  : qb;
    const float scale         = blockIdx.y ? 1.0f : QSCALE;

    int wave = threadIdx.x >> 5;
    int lane = threadIdx.x & 31;
    int half = lane >> 4;
    int mcol = lane & 15;

    int t = blockIdx.x * 8 + wave;            // 8192 tiles total
    int rowTile = t >> 4;                     // 0..511  (B*N/32 row tiles)
    int colBase = (t & 15) << 6;              // 0..960 step 64

    const bf16_t* A0base = (const bf16_t*)xb + (size_t)(rowTile * 32 + mcol) * Dd;
    const bf16_t* A1base = A0base + (size_t)16 * Dd;
    const bf16_t* Wbase  = (const bf16_t*)W;

    auto loadA = [&](const bf16_t* base, int kbd) {
        return ld16(base + kbd + half * 8, base + kbd + 16 + half * 8);
    };
    auto loadB = [&](int nt, int kbd) {
        const bf16_t* p = Wbase + (size_t)(colBase + nt * 16 + mcol) * Dd + kbd + half * 8;
        return ld16(p, p + 16);
    };

    v8f zero = {0.f, 0.f, 0.f, 0.f, 0.f, 0.f, 0.f, 0.f};
    v8f c[8];
#pragma unroll
    for (int i = 0; i < 8; ++i) c[i] = zero;

    v16bf a0 = loadA(A0base, 0), a1 = loadA(A1base, 0);
    v16bf b0 = loadB(0, 0), b1 = loadB(1, 0), b2 = loadB(2, 0), b3 = loadB(3, 0);

    for (int kbd = 0; kbd < Dd; kbd += 64) {
        int k1 = kbd + 32;
        v16bf na0 = loadA(A0base, k1), na1 = loadA(A1base, k1);
        v16bf nb0 = loadB(0, k1), nb1 = loadB(1, k1), nb2 = loadB(2, k1), nb3 = loadB(3, k1);

        c[0] = wmma_bf16(a0, b0, c[0]);
        c[1] = wmma_bf16(a0, b1, c[1]);
        c[2] = wmma_bf16(a0, b2, c[2]);
        c[3] = wmma_bf16(a0, b3, c[3]);
        c[4] = wmma_bf16(a1, b0, c[4]);
        c[5] = wmma_bf16(a1, b1, c[5]);
        c[6] = wmma_bf16(a1, b2, c[6]);
        c[7] = wmma_bf16(a1, b3, c[7]);

        int k2 = (kbd + 64) & (Dd - 1);       // wraps to 0 on last iter (dummy)
        a0 = loadA(A0base, k2); a1 = loadA(A1base, k2);
        b0 = loadB(0, k2); b1 = loadB(1, k2); b2 = loadB(2, k2); b3 = loadB(3, k2);

        c[0] = wmma_bf16(na0, nb0, c[0]);
        c[1] = wmma_bf16(na0, nb1, c[1]);
        c[2] = wmma_bf16(na0, nb2, c[2]);
        c[3] = wmma_bf16(na0, nb3, c[3]);
        c[4] = wmma_bf16(na1, nb0, c[4]);
        c[5] = wmma_bf16(na1, nb1, c[5]);
        c[6] = wmma_bf16(na1, nb2, c[6]);
        c[7] = wmma_bf16(na1, nb3, c[7]);
    }

#pragma unroll
    for (int mt = 0; mt < 2; ++mt) {
#pragma unroll
        for (int nt = 0; nt < 4; ++nt) {
#pragma unroll
            for (int v = 0; v < 8; ++v) {
                int r = rowTile * 32 + mt * 16 + v + 8 * half;
                int col = colBase + nt * 16 + mcol;
                dst[(size_t)r * Dd + col] = f32_to_bf16_bits(c[mt * 4 + nt][v] * scale);
            }
        }
    }
}

// ---------------- flash attention: softmax(q k^T) x ----------------
// 512 threads = 16 waves = 2 query-groups x 8 D-slices; 32 query rows per block.
// k/x tiles staged in LDS with async-to-LDS double buffering (one j0 ahead).
__global__ __launch_bounds__(512) void attn_kernel(const unsigned short* __restrict__ qb,
                                                   const unsigned short* __restrict__ kb,
                                                   const unsigned short* __restrict__ xt,
                                                   float* __restrict__ out) {
    extern __shared__ char smem[];
    float* S_lds = (float*)(smem + 2 * KTILE_B + 2 * XTILE_B);   // [2][16*32]
    float* l_lds = S_lds + 1024;                                  // [2][16]
    float* a_lds = l_lds + 32;                                    // [2][16]

    int tid  = threadIdx.x;
    int wave = tid >> 5;
    int lane = tid & 31;
    int half = lane >> 4;
    int mcol = lane & 15;
    int g    = wave >> 3;              // query group 0/1
    int ds   = wave & 7;               // D-slice 0..7
    int dsBase = ds * 128;

    int b   = blockIdx.x >> 7;         // batch
    int qbi = blockIdx.x & 127;        // 32-row query block
    int r0  = qbi * 32 + g * 16;       // this group's first query row
    size_t batchOff = (size_t)b * Nn * Dd;

    const bf16_t* kbb = (const bf16_t*)kb + batchOff;
    const bf16_t* xtb = (const bf16_t*)xt + (size_t)b * Dd * Nn;
    const bf16_t* qrow = (const bf16_t*)qb + batchOff + (size_t)(r0 + mcol) * Dd + dsBase;

    // buffer base pointers computed from runtime index (no static pointer tables)
    auto kBuf = [&](int bs) { return (bf16_t*)(smem + bs * KTILE_B); };
    auto xBuf = [&](int bs) { return (bf16_t*)(smem + 2 * KTILE_B + bs * XTILE_B); };

    // stage one 32-key tile (k rows + x^T rows) into LDS buffer `bs`
    auto stage = [&](int j0, int bs) {
        bf16_t* kd = kBuf(bs);
        bf16_t* xd = xBuf(bs);
#pragma unroll
        for (int i = 0; i < 8; ++i) {           // k: 32 keys x 1024 dims, 4096 16B chunks
            int c = tid + i * 512;
            int key = c >> 7, dc = c & 127;
            async_copy16(kbb + (size_t)(j0 + key) * Dd + dc * 8, kd + key * KROW + dc * 8);
        }
#pragma unroll
        for (int i = 0; i < 8; ++i) {           // x^T: 1024 dims x 32 keys, 4096 16B chunks
            int c = tid + i * 512;
            int dim = c >> 2, kc = c & 3;
            async_copy16(xtb + (size_t)dim * Nn + j0 + kc * 8, xd + dim * XROW + kc * 8);
        }
    };

    // q fragments: loop-invariant
    v16bf qf[4];
#pragma unroll
    for (int kk = 0; kk < 4; ++kk)
        qf[kk] = ld16(qrow + kk * 32 + half * 8, qrow + kk * 32 + 16 + half * 8);

    v8f zero = {0.f, 0.f, 0.f, 0.f, 0.f, 0.f, 0.f, 0.f};
    v8f o[8];
#pragma unroll
    for (int t = 0; t < 8; ++t) o[t] = zero;

    float mprev = -3.0e38f;            // private running max for row = mcol (same in both halves)
    if (tid < 32) l_lds[tid] = 0.f;

    stage(0, 0);
    async_wait_all();
    __syncthreads();

    int buf = 0;
    for (int j0 = 0; j0 < Nn; j0 += 32) {
        if (j0 + 32 < Nn) stage(j0 + 32, buf ^ 1);   // async prefetch next tile

        // ---- zero score tiles ----
        S_lds[tid] = 0.f;
        S_lds[tid + 512] = 0.f;
        __syncthreads();

        // ---- partial S over this wave's 128 dims, from LDS k tile (pipelined) ----
        const bf16_t* kcur = kBuf(buf);
        auto loadK = [&](int kk, int sub) {
            const bf16_t* p = kcur + (size_t)(sub * 16 + mcol) * KROW + dsBase + kk * 32 + half * 8;
            return ld16(p, p + 16);
        };
        v8f s0 = zero, s1 = zero;
        v16bf k0a = loadK(0, 0), k1a = loadK(0, 1);
        v16bf k0b = loadK(1, 0), k1b = loadK(1, 1);
        s0 = wmma_bf16(qf[0], k0a, s0);
        s1 = wmma_bf16(qf[0], k1a, s1);
        k0a = loadK(2, 0); k1a = loadK(2, 1);
        s0 = wmma_bf16(qf[1], k0b, s0);
        s1 = wmma_bf16(qf[1], k1b, s1);
        k0b = loadK(3, 0); k1b = loadK(3, 1);
        s0 = wmma_bf16(qf[2], k0a, s0);
        s1 = wmma_bf16(qf[2], k1a, s1);
        s0 = wmma_bf16(qf[3], k0b, s0);
        s1 = wmma_bf16(qf[3], k1b, s1);

        float* Sg = S_lds + g * 512;
#pragma unroll
        for (int v = 0; v < 8; ++v) {
            int rr = v + 8 * half;
            atomicAdd(&Sg[rr * 32 + mcol], s0[v]);
            atomicAdd(&Sg[rr * 32 + 16 + mcol], s1[v]);
        }
        __syncthreads();

        // ---- online softmax: all waves of a group compute redundantly in parallel;
        //      P built directly in registers in A-fragment halfword order ----
        {
            int row = mcol;
            float sv[16];
            float tmax = -3.0e38f;
#pragma unroll
            for (int j = 0; j < 16; ++j) {
                int col = j + half * 8 + ((j >= 8) ? 8 : 0);
                sv[j] = Sg[row * 32 + col];
                tmax = fmaxf(tmax, sv[j]);
            }
            float pmax = __shfl_xor(tmax, 16, 32);
            tmax = fmaxf(tmax, pmax);
            float mn = fmaxf(mprev, tmax);
            float al = __expf(mprev - mn);
            mprev = mn;
            float ps = 0.f;
            union { v16bf v; unsigned int u[8]; } pu;
#pragma unroll
            for (int jj = 0; jj < 8; ++jj) {
                float p0 = __expf(sv[2 * jj] - mn);
                float p1 = __expf(sv[2 * jj + 1] - mn);
                ps += p0 + p1;
                pu.u[jj] = (unsigned int)f32_to_bf16_bits(p0) |
                           ((unsigned int)f32_to_bf16_bits(p1) << 16);
            }
            ps += __shfl_xor(ps, 16, 32);
            if (ds == 0 && half == 0) {
                a_lds[g * 16 + row] = al;
                l_lds[g * 16 + row] = l_lds[g * 16 + row] * al + ps;
            }
            __syncthreads();

            // ---- rescale O and accumulate P * x_slice from LDS x tile (pipelined) ----
            float av[8];
#pragma unroll
            for (int v = 0; v < 8; ++v) av[v] = a_lds[g * 16 + v + 8 * half];
#pragma unroll
            for (int t = 0; t < 8; ++t) {
#pragma unroll
                for (int v = 0; v < 8; ++v) o[t][v] *= av[v];
            }

            const bf16_t* xcur = xBuf(buf);
            auto loadX = [&](int t) {
                const bf16_t* p = xcur + (size_t)(dsBase + t * 16 + mcol) * XROW + half * 8;
                return ld16(p, p + 16);
            };
            v16bf xf[2];
            xf[0] = loadX(0);
#pragma unroll
            for (int t = 0; t < 8; ++t) {
                if (t < 7) xf[(t + 1) & 1] = loadX(t + 1);
                o[t] = wmma_bf16(pu.v, xf[t & 1], o[t]);
            }
        }

        async_wait_all();      // next tile staged
        __syncthreads();       // everyone done with this buffer + next tile visible
        buf ^= 1;
    }

    // ---- normalize and store f32 output ----
    float lv[8];
#pragma unroll
    for (int v = 0; v < 8; ++v) lv[v] = 1.0f / l_lds[g * 16 + v + 8 * half];
#pragma unroll
    for (int t = 0; t < 8; ++t) {
#pragma unroll
        for (int v = 0; v < 8; ++v) {
            size_t idx = batchOff + (size_t)(r0 + v + 8 * half) * Dd + dsBase + t * 16 + mcol;
            out[idx] = o[t][v] * lv[v];
        }
    }
}

extern "C" void kernel_launch(void* const* d_in, const int* in_sizes, int n_in,
                              void* d_out, int out_size, void* d_ws, size_t ws_size,
                              hipStream_t stream) {
    const float* x  = (const float*)d_in[0];
    const float* Wq = (const float*)d_in[1];
    const float* Wk = (const float*)d_in[2];
    float* out = (float*)d_out;

    const size_t XN = (size_t)Bq * Nn * Dd;   // 16777216
    const size_t WN = (size_t)Dd * Dd;        // 1048576

    unsigned short* qb  = (unsigned short*)d_ws;
    unsigned short* kb  = qb  + XN;
    unsigned short* xb  = kb  + XN;
    unsigned short* xt  = xb  + XN;
    unsigned short* wqb = xt  + XN;
    unsigned short* wkb = wqb + WN;

    cvt_x_kernel<<<(unsigned)(XN / 256), 256, 0, stream>>>(x, xb, xt);
    cvt_w_kernel<<<(unsigned)(WN / 256), 256, 0, stream>>>(Wq, Wk, wqb, wkb);

    proj_kernel<<<dim3(1024, 2), 256, 0, stream>>>(xb, wqb, wkb, qb, kb);

    // one block per (batch, 32-query tile): 4*128 = 512 blocks, 512 threads, ~300KB LDS
    (void)hipFuncSetAttribute((const void*)attn_kernel,
                              hipFuncAttributeMaxDynamicSharedMemorySize, SMEM_BYTES);
    attn_kernel<<<512, 512, SMEM_BYTES, stream>>>(qb, kb, xt, out);
}
